// ResidualAttentionBlock_84447646974224
// MI455X (gfx1250) — matrix operand order repaired and checked by reading, not compiled, not run on hardware
//
#include <hip/hip_runtime.h>
#include <hip/hip_bf16.h>
#include <math.h>

// ---------------------------------------------------------------------------
// Problem constants (from reference)
// ---------------------------------------------------------------------------
#define N_NODES 50000
#define N_EDGES 800000
#define HEADS   10
#define DHEAD   13
#define FDIM    128        // IN_F == OUT_F
#define HD      130        // HEADS*DHEAD
#define HD_PAD  160        // HD padded to multiple of 32 (WMMA K)

typedef __attribute__((ext_vector_type(16))) __bf16 v16bf;
typedef __attribute__((ext_vector_type(8)))  float  v8f;

// ---------------------------------------------------------------------------
// Helpers
// ---------------------------------------------------------------------------
__device__ __forceinline__ unsigned short f32_to_bf16(float f) {
  unsigned u = __float_as_uint(f);
  unsigned r = u + 0x7FFFu + ((u >> 16) & 1u);   // round-to-nearest-even
  return (unsigned short)(r >> 16);
}

// order-preserving float<->uint for atomicMax-based segment max
__device__ __forceinline__ unsigned fenc(float f) {
  int i = __float_as_int(f);
  return (unsigned)(i >= 0 ? (i | (int)0x80000000) : ~i);
}
__device__ __forceinline__ float fdec(unsigned u) {
  int i = (u & 0x80000000u) ? (int)(u & 0x7FFFFFFFu) : ~(int)u;
  return __int_as_float(i);
}

__device__ __forceinline__ float elu(float x) {
  return x > 0.f ? x : (__expf(x) - 1.f);
}

// CDNA5 async copy: global -> LDS, 16B per lane, tracked by ASYNCcnt.
// ldsAddr = LDS byte offset (low 32 bits of generic pointer to __shared__).
__device__ __forceinline__ void async_ld_b128(unsigned ldsAddr, const void* g) {
  unsigned long long ga = (unsigned long long)g;
  asm volatile("global_load_async_to_lds_b128 %0, %1, off"
               :: "v"(ldsAddr), "v"(ga)
               : "memory");
}
__device__ __forceinline__ void wait_asynccnt0() {
  asm volatile("s_wait_asynccnt 0x0" ::: "memory");
}

// ---------------------------------------------------------------------------
// LayerNorm (one wave32 per 128-wide row) -> f32 out (skip path) + bf16 out
// ---------------------------------------------------------------------------
__global__ void ln_kernel(const float* __restrict__ x,
                          const float* __restrict__ w,
                          const float* __restrict__ b,
                          float* __restrict__ outF,
                          unsigned short* __restrict__ outB, int n) {
  int wave = (int)((blockIdx.x * blockDim.x + threadIdx.x) >> 5);
  int lane = threadIdx.x & 31;
  if (wave >= n) return;
  const float* row = x + (size_t)wave * FDIM;
  float v[4];
  float s = 0.f, ss = 0.f;
#pragma unroll
  for (int j = 0; j < 4; ++j) {
    v[j] = row[lane + 32 * j];
    s += v[j];
    ss += v[j] * v[j];
  }
#pragma unroll
  for (int off = 16; off; off >>= 1) {
    s  += __shfl_xor(s,  off, 32);
    ss += __shfl_xor(ss, off, 32);
  }
  float mu   = s * (1.f / FDIM);
  float var  = ss * (1.f / FDIM) - mu * mu;
  float rstd = rsqrtf(var + 1e-5f);
#pragma unroll
  for (int j = 0; j < 4; ++j) {
    int c = lane + 32 * j;
    float y = (v[j] - mu) * rstd * w[c] + b[c];
    if (outF) outF[(size_t)wave * FDIM + c] = y;
    outB[(size_t)wave * FDIM + c] = f32_to_bf16(y);
  }
}

// ---------------------------------------------------------------------------
// Pack fp32 weight [Ksrc x Nsrc] row-major into bf16 WMMA B-operand fragments.
// Fragment (kt, nt): 32 lanes x 8 dwords; lane&15 = column, lane>>4 = K-half,
// dword v holds K = kt*32 + 2v + 16*(lane>>4) and K+1 (pair-packed bf16).
// ---------------------------------------------------------------------------
__global__ void pack_b_kernel(const float* __restrict__ src,
                              unsigned* __restrict__ dst,
                              int Ksrc, int Nsrc, int KT, int NT) {
  int t = blockIdx.x * blockDim.x + threadIdx.x;
  int total = KT * NT * 256;
  if (t >= total) return;
  int kt   = t / (NT * 256);
  int r    = t % (NT * 256);
  int nt   = r / 256;
  int q    = r % 256;
  int lane = q >> 3;
  int v    = q & 7;
  int khalf = lane >> 4;
  int n = nt * 16 + (lane & 15);
  int k = kt * 32 + 2 * v + 16 * khalf;
  float a0 = (k     < Ksrc && n < Nsrc) ? src[(size_t)k * Nsrc + n]       : 0.f;
  float a1 = (k + 1 < Ksrc && n < Nsrc) ? src[(size_t)(k + 1) * Nsrc + n] : 0.f;
  dst[t] = (unsigned)f32_to_bf16(a0) | ((unsigned)f32_to_bf16(a1) << 16);
}

// ---------------------------------------------------------------------------
// Wave GEMM with LDS-staged B:
//  - whole block async-copies a group of packed B fragments into LDS
//    (global_load_async_to_lds_b128 + s_wait_asynccnt), then
//  - each wave computes a 16-row M-tile with v_wmma_f32_16x16x32_bf16,
//    reading B via ds_load_b128.
// A: bf16 [M, KTILES*32] row-major. Out row-major ldc=NCOLS.
// ---------------------------------------------------------------------------
template <int NCOLS, int KTILES, bool BIAS, bool ELU_, bool SKIP, bool OUT_BF16>
__global__ void wave_gemm(const unsigned short* __restrict__ A,
                          const unsigned* __restrict__ Bp,
                          const float* __restrict__ bias,
                          const float* __restrict__ skip,
                          float* __restrict__ outF,
                          unsigned short* __restrict__ outB, int M) {
  constexpr int NT = (NCOLS + 15) / 16;
  constexpr int K  = KTILES * 32;
  // group of n-tiles staged at once; cap LDS at 64KB (KTILES*G KB)
  constexpr int G  = (64 / KTILES) < NT ? (64 / KTILES) : NT;

  __shared__ uint4 shB[KTILES * G * 64];   // KTILES*G fragments, 1KB each
  const unsigned shBase = (unsigned)(unsigned long long)(void*)shB;

  int wave = (int)((blockIdx.x * blockDim.x + threadIdx.x) >> 5);
  int lane = threadIdx.x & 31;
  int tid  = threadIdx.x;
  int m0 = wave * 16;
  if (m0 + 16 > M) m0 = (M > 16) ? (M - 16) : 0;  // clamp: keep all waves in barriers
  int khalf = lane >> 4;
  int ml    = lane & 15;

  union Frag { uint4 u[2]; v16bf v; };
  // A fragments: lane ml holds row m0+ml; khalf selects K 0..15 / 16..31 half.
  Frag a[KTILES];
  const unsigned short* Arow = A + (size_t)(m0 + ml) * K;
#pragma unroll
  for (int kt = 0; kt < KTILES; ++kt) {
    a[kt].u[0] = *(const uint4*)(Arow + kt * 32 + khalf * 16);       // K k0..k0+7
    a[kt].u[1] = *(const uint4*)(Arow + kt * 32 + khalf * 16 + 8);   // K k0+8..k0+15
  }

  int rbase = m0 + (khalf ? 8 : 0);

  for (int g0 = 0; g0 < NT; g0 += G) {
    const int gcnt = (NT - g0) < G ? (NT - g0) : G;
    __syncthreads();   // previous group's LDS reads complete before overwrite
    // --- async stage: fragments (kt, g0..g0+gcnt-1) -> LDS ---
    {
      const int units = KTILES * gcnt * 64;  // 16B units
      for (int u = tid; u < units; u += 256) {
        int kt  = u / (gcnt * 64);
        int rem = u - kt * (gcnt * 64);
        unsigned lds = shBase + (unsigned)((kt * (G * 64) + rem) * 16);
        const char* g = (const char*)Bp + ((size_t)(kt * NT + g0) * 64 + rem) * 16;
        async_ld_b128(lds, g);
      }
      wait_asynccnt0();
    }
    __syncthreads();

    // --- compute staged n-tiles ---
    for (int gi = 0; gi < gcnt; ++gi) {
      v8f acc = {};
#pragma unroll
      for (int kt = 0; kt < KTILES; ++kt) {
        Frag b;
        const uint4* bf = shB + (kt * G + gi) * 64;
        b.u[0] = bf[lane * 2];
        b.u[1] = bf[lane * 2 + 1];
        acc = __builtin_amdgcn_wmma_f32_16x16x32_bf16(false, a[kt].v, false, b.v,
                                                      (short)0, acc, false, false);
      }
      int n = (g0 + gi) * 16 + ml;
      if (n < NCOLS) {
        float bv = 0.f;
        if constexpr (BIAS) bv = bias[n];
#pragma unroll
        for (int v = 0; v < 8; ++v) {
          float x = acc[v] + bv;
          if constexpr (ELU_) x = elu(x);
          int row = rbase + v;
          size_t idx = (size_t)row * NCOLS + n;
          if constexpr (SKIP) x += skip[idx];
          if constexpr (OUT_BF16) outB[idx] = f32_to_bf16(x);
          else                    outF[idx] = x;
        }
      }
    }
  }
}

// ---------------------------------------------------------------------------
// Edge kernels: scores + segment-max, then exp / segment-sum / aggregation
// ---------------------------------------------------------------------------
__global__ void edge_scores_kernel(const float* __restrict__ ft,
                                   const int* __restrict__ src,
                                   const int* __restrict__ dst,
                                   float* __restrict__ sc,
                                   unsigned* __restrict__ mEnc) {
  int t = blockIdx.x * blockDim.x + threadIdx.x;
  if (t >= N_EDGES * HEADS) return;
  int e = t / HEADS, h = t % HEADS;
  int s = src[e], d = dst[e];
  const float* fs = ft + (size_t)s * HD + h * DHEAD;
  const float* fd = ft + (size_t)d * HD + h * DHEAD;
  float dot = 0.f;
#pragma unroll
  for (int i = 0; i < DHEAD; ++i) dot += fs[i] * fd[i];
  float score = dot * 0.2773500981126146f;   // 1/sqrt(13)
  sc[t] = score;
  atomicMax(mEnc + (size_t)d * HEADS + h, fenc(score));
}

__global__ void edge_agg_kernel(const float* __restrict__ ft,
                                const int* __restrict__ src,
                                const int* __restrict__ dst,
                                const unsigned* __restrict__ mEnc,
                                const float* __restrict__ sc,
                                float* __restrict__ z,
                                float* __restrict__ agg) {
  int t = blockIdx.x * blockDim.x + threadIdx.x;
  if (t >= N_EDGES * HEADS) return;
  int e = t / HEADS, h = t % HEADS;
  int s = src[e], d = dst[e];
  float mm = fdec(mEnc[(size_t)d * HEADS + h]);
  float ex = __expf(sc[t] - mm);
  atomicAdd(z + (size_t)d * HEADS + h, ex);
  const float* fs = ft + (size_t)s * HD + h * DHEAD;
  float* ag = agg + (size_t)d * HD + h * DHEAD;
#pragma unroll
  for (int i = 0; i < DHEAD; ++i) atomicAdd(ag + i, fs[i] * ex);
}

// agg/z -> ELU -> bf16, padded [N, 160] (zeros beyond col 130)
__global__ void act_kernel(const float* __restrict__ agg,
                           const float* __restrict__ z,
                           unsigned short* __restrict__ act) {
  int t = blockIdx.x * blockDim.x + threadIdx.x;
  if (t >= N_NODES * HD_PAD) return;
  int nrow = t / HD_PAD, c = t % HD_PAD;
  float v = 0.f;
  if (c < HD) {
    int h = c / DHEAD;
    float zz = z[(size_t)nrow * HEADS + h];
    float a  = agg[(size_t)nrow * HD + c];
    v = (zz > 0.f) ? (a / zz) : 0.f;
    v = elu(v);
  }
  act[t] = f32_to_bf16(v);
}

// ---------------------------------------------------------------------------
// Launch
// ---------------------------------------------------------------------------
extern "C" void kernel_launch(void* const* d_in, const int* in_sizes, int n_in,
                              void* d_out, int out_size, void* d_ws, size_t ws_size,
                              hipStream_t stream) {
  const float* h     = (const float*)d_in[0];
  const int*   src   = (const int*)d_in[1];
  const int*   dst   = (const int*)d_in[2];
  const float* ln1_w = (const float*)d_in[3];
  const float* ln1_b = (const float*)d_in[4];
  const float* W_att = (const float*)d_in[5];
  const float* hr_w  = (const float*)d_in[6];
  const float* hr_b  = (const float*)d_in[7];
  const float* in_w  = (const float*)d_in[8];
  const float* in_b  = (const float*)d_in[9];
  const float* w1    = (const float*)d_in[10];
  const float* b1    = (const float*)d_in[11];
  const float* w2    = (const float*)d_in[12];
  const float* b2    = (const float*)d_in[13];

  // --- workspace carve (256B aligned) ---
  char* ws = (char*)d_ws;
  size_t off = 0;
  auto carve = [&](size_t bytes) -> void* {
    off = (off + 255) & ~(size_t)255;
    void* p = ws + off;
    off += bytes;
    return p;
  };
  float*          hn      = (float*)carve((size_t)N_NODES * FDIM * 4);
  unsigned short* hn_bf   = (unsigned short*)carve((size_t)N_NODES * FDIM * 2);
  float*          ft      = (float*)carve((size_t)N_NODES * HD * 4);
  float*          sc      = (float*)carve((size_t)N_EDGES * HEADS * 4);
  unsigned*       mEnc    = (unsigned*)carve((size_t)N_NODES * HEADS * 4);
  float*          z       = (float*)carve((size_t)N_NODES * HEADS * 4);
  float*          agg     = (float*)carve((size_t)N_NODES * HD * 4);
  unsigned short* act1    = (unsigned short*)carve((size_t)N_NODES * HD_PAD * 2);
  float*          out2    = (float*)carve((size_t)N_NODES * FDIM * 4);
  float*          skip2   = (float*)carve((size_t)N_NODES * FDIM * 4);
  unsigned short* x2      = (unsigned short*)carve((size_t)N_NODES * FDIM * 2);
  unsigned short* tbuf    = (unsigned short*)carve((size_t)N_NODES * 4 * FDIM * 2);
  unsigned*       BpAtt   = (unsigned*)carve((size_t)4  * 9  * 256 * 4);  // K=128, N=130
  unsigned*       BpHr    = (unsigned*)carve((size_t)5  * 8  * 256 * 4);  // K=160, N=128
  unsigned*       BpW1    = (unsigned*)carve((size_t)4  * 32 * 256 * 4);  // K=128, N=512
  unsigned*       BpW2    = (unsigned*)carve((size_t)16 * 8  * 256 * 4);  // K=512, N=128

  // --- zero/init accumulators (graph-capture safe) ---
  hipMemsetAsync(mEnc, 0, (size_t)N_NODES * HEADS * 4, stream);  // fenc(-max)
  hipMemsetAsync(z,    0, (size_t)N_NODES * HEADS * 4, stream);
  hipMemsetAsync(agg,  0, (size_t)N_NODES * HD * 4, stream);

  // --- pack weights into WMMA B-fragment layout (bf16) ---
  {
    int t;
    t = 4 * 9 * 256;
    pack_b_kernel<<<(t + 255) / 256, 256, 0, stream>>>(W_att, BpAtt, 128, 130, 4, 9);
    t = 5 * 8 * 256;
    pack_b_kernel<<<(t + 255) / 256, 256, 0, stream>>>(hr_w, BpHr, 130, 128, 5, 8);
    t = 4 * 32 * 256;
    pack_b_kernel<<<(t + 255) / 256, 256, 0, stream>>>(w1, BpW1, 128, 512, 4, 32);
    t = 16 * 8 * 256;
    pack_b_kernel<<<(t + 255) / 256, 256, 0, stream>>>(w2, BpW2, 512, 128, 16, 8);
  }

  const int lnBlocks   = (N_NODES * 32 + 255) / 256;     // one wave per row
  const int gemmBlocks = ((N_NODES / 16) * 32 + 255) / 256;
  const int edgeBlocks = (N_EDGES * HEADS + 255) / 256;
  const int actBlocks  = (N_NODES * HD_PAD + 255) / 256;

  // 1) hn = LN(h) -> f32 (skip) + bf16 (WMMA A)
  ln_kernel<<<lnBlocks, 256, 0, stream>>>(h, ln1_w, ln1_b, hn, hn_bf, N_NODES);

  // 2) ft = hn @ W_att   [N,130] f32
  wave_gemm<130, 4, false, false, false, false><<<gemmBlocks, 256, 0, stream>>>(
      hn_bf, BpAtt, nullptr, nullptr, ft, nullptr, N_NODES);

  // 3) per-(edge,head) scores + segment max
  edge_scores_kernel<<<edgeBlocks, 256, 0, stream>>>(ft, src, dst, sc, mEnc);

  // 4) exp, segment sum Z, weighted aggregation (division by Z deferred)
  edge_agg_kernel<<<edgeBlocks, 256, 0, stream>>>(ft, src, dst, mEnc, sc, z, agg);

  // 5) elu(agg / Z) -> bf16 padded [N,160]
  act_kernel<<<actBlocks, 256, 0, stream>>>(agg, z, act1);

  // 6) out2 = act1 @ hr_w + hr_b + hn   [N,128] f32
  wave_gemm<128, 5, true, false, true, false><<<gemmBlocks, 256, 0, stream>>>(
      act1, BpHr, hr_b, hn, out2, nullptr, N_NODES);

  // 7) skip2/x2 = LN(out2)
  ln_kernel<<<lnBlocks, 256, 0, stream>>>(out2, in_w, in_b, skip2, x2, N_NODES);

  // 8) t = elu(x2 @ w1 + b1)   [N,512] bf16
  wave_gemm<512, 4, true, true, false, true><<<gemmBlocks, 256, 0, stream>>>(
      x2, BpW1, b1, nullptr, nullptr, tbuf, N_NODES);

  // 9) out = elu(t @ w2 + b2) + skip2   [N,128] f32 -> d_out
  wave_gemm<128, 16, true, true, true, false><<<gemmBlocks, 256, 0, stream>>>(
      tbuf, BpW2, b2, skip2, (float*)d_out, nullptr, N_NODES);

  (void)in_sizes; (void)n_in; (void)out_size; (void)ws_size;
}
